// ScalableSpikingAttention_87454124081927
// MI455X (gfx1250) — matrix operand order repaired and checked by reading, not compiled, not run on hardware
//
#include <hip/hip_runtime.h>
#include <hip/hip_bf16.h>
#include <cmath>

// ---- problem constants (match reference) ----
#define BB   32
#define NQ   1024
#define NK   2048
#define DH   64

#define QTILE      16               // queries per workgroup
#define NWAVES     8                // 256 threads
#define KT1        16               // key tile, phase 1 (scores)
#define KT3        32               // key tile, phase 3 (attn @ V)

#define DECAY_F     0.95122942450071400f      // exp(-1/20)
#define TWO_PI_F    6.28318530717958647f
#define INV_2PW2_F  0.81056946913870217f      // 1/(2*(2pi/8)^2) = 8/pi^2... see below
// 2*PHASE_WIDTH^2 = 2*(pi/4)^2 = pi^2/8 = 1.2337005501361697 -> inverse:
#undef  INV_2PW2_F
#define INV_2PW2_F  0.81056946913870217f
#define SCALE_F     0.125f                    // 1/sqrt(64)

// ---- LDS layout (dynamic) ----
#define SC_BYTES     (QTILE * NK * 4)          // 131072 : f32 scores/probs
#define QT_BYTES     (QTILE * DH * 2)          //   2048 : f16 q traces
#define STAGE_BYTES  (NWAVES * 4096)           //  32768 : per-wave f16 staging
#define OACC_BYTES   (QTILE * DH * 4)          //   4096 : f32 output accumulator
#define SMEM_TOTAL   (SC_BYTES + QT_BYTES + STAGE_BYTES + OACC_BYTES)  // 169984

typedef _Float16 v16h __attribute__((ext_vector_type(16)));
typedef _Float16 v8h  __attribute__((ext_vector_type(8)));
typedef _Float16 v4h  __attribute__((ext_vector_type(4)));
typedef float    v8f  __attribute__((ext_vector_type(8)));

static __device__ __forceinline__ v16h pack2x8(const _Float16* p0, const _Float16* p1) {
    v8h a = *(const v8h*)p0;
    v8h b = *(const v8h*)p1;
    return __builtin_shufflevector(a, b, 0,1,2,3,4,5,6,7,8,9,10,11,12,13,14,15);
}

static __device__ __forceinline__ v16h cvt2x8(const float* p0, const float* p1) {
    v16h r;
#pragma unroll
    for (int i = 0; i < 8; ++i) {
        r[i]     = (_Float16)p0[i];
        r[i + 8] = (_Float16)p1[i];
    }
    return r;
}

static __device__ __forceinline__ void lds_wait() {
    asm volatile("s_wait_dscnt 0x0" ::: "memory");
}

__global__ __launch_bounds__(256, 1)
void spiking_attn_wmma_kernel(const float* __restrict__ query_spikes,
                              const float* __restrict__ key_spikes,
                              const float* __restrict__ value,
                              const float* __restrict__ query_trace,
                              const float* __restrict__ key_trace,
                              const float* __restrict__ key_phases,
                              const float* __restrict__ current_phase,
                              float* __restrict__ out,       // [B,NQ,DH]
                              float* __restrict__ attn) {    // [B,NQ,NK]
    extern __shared__ __align__(128) char smem[];
    float*    sc    = (float*)smem;                               // [QTILE][NK]
    _Float16* qt    = (_Float16*)(smem + SC_BYTES);               // [QTILE][DH]
    _Float16* stage = (_Float16*)(smem + SC_BYTES + QT_BYTES);    // per-wave 2048 halves
    float*    oacc  = (float*)(smem + SC_BYTES + QT_BYTES + STAGE_BYTES); // [QTILE][DH]

    const int t    = threadIdx.x;
    const int lane = t & 31;
    const int wv   = t >> 5;
    const int b    = blockIdx.x / (NQ / QTILE);
    const int q0   = (blockIdx.x % (NQ / QTILE)) * QTILE;

    // ---------------- prologue: q_tr tile -> f16 LDS ----------------
    {
        const size_t base = ((size_t)b * NQ + q0) * DH;           // 16*64 contiguous f32
        const float4* qs4 = (const float4*)(query_spikes + base);
        const float4* tr4 = (const float4*)(query_trace + base);
        float4 s = qs4[t];
        float4 r = tr4[t];
        v4h h = { (_Float16)fmaf(DECAY_F, r.x, s.x),
                  (_Float16)fmaf(DECAY_F, r.y, s.y),
                  (_Float16)fmaf(DECAY_F, r.z, s.z),
                  (_Float16)fmaf(DECAY_F, r.w, s.w) };
        *(v4h*)(qt + t * 4) = h;
    }
    __syncthreads();

    const float cp = current_phase[0];
    const int   m   = lane & 15;       // row / N index inside 16-wide tiles
    const int   hi  = lane >> 4;       // half-wave select
    const int   kb0 = hi * 16;         // A-operand K-chunk base

    // hoisted A operands for Q (16x32 f16 A-tile layout, two K=32 slabs)
    const v16h qa0 = pack2x8(qt + m * DH + kb0,      qt + m * DH + 8 + kb0);
    const v16h qa1 = pack2x8(qt + m * DH + 32 + kb0, qt + m * DH + 40 + kb0);

    _Float16* wstage = stage + wv * 2048;

    // ---------------- phase 1: scores = gate * (q_tr . k_tr^T) / 8 ----------------
    for (int kt = wv; kt < NK / KT1; kt += NWAVES) {
        const int kbase = kt * KT1;

        // stage k_tr tile (16 keys x 64 dh) as f16
        {
            const int key = lane >> 1;
            const int d0  = (lane & 1) * 32;
            const size_t goff = ((size_t)b * NK + kbase + key) * DH + d0;
            const float4* s4 = (const float4*)(key_spikes + goff);
            const float4* r4 = (const float4*)(key_trace + goff);
            // prefetch next tile's key data into cache
            __builtin_prefetch(key_spikes + goff + (size_t)NWAVES * KT1 * DH, 0, 1);
            _Float16* dst = wstage + key * DH + d0;
#pragma unroll
            for (int jj = 0; jj < 8; ++jj) {
                float4 s = s4[jj];
                float4 r = r4[jj];
                v4h h = { (_Float16)fmaf(DECAY_F, r.x, s.x),
                          (_Float16)fmaf(DECAY_F, r.y, s.y),
                          (_Float16)fmaf(DECAY_F, r.z, s.z),
                          (_Float16)fmaf(DECAY_F, r.w, s.w) };
                *(v4h*)(dst + jj * 4) = h;
            }
        }
        lds_wait();

        // B operand: lane covers key column n=m, K (=dh) contiguous
        const int koff = hi * 16;
        const v16h b0 = *(const v16h*)(wstage + m * DH + koff);
        const v16h b1 = *(const v16h*)(wstage + m * DH + 32 + koff);

        v8f c = {};
        c = __builtin_amdgcn_wmma_f32_16x16x32_f16(false, qa0, false, b0, (short)0, c, false, false);
        c = __builtin_amdgcn_wmma_f32_16x16x32_f16(false, qa1, false, b1, (short)0, c, false, false);

        // phase gate folded with 1/sqrt(DH); depends only on this lane's key column
        float kp = key_phases[kbase + m];
        float pd = fabsf(kp - cp);
        pd = fminf(pd, TWO_PI_F - pd);
        const float g = SCALE_F * __expf(-pd * pd * INV_2PW2_F);

#pragma unroll
        for (int r = 0; r < 8; ++r)
            sc[(r + hi * 8) * NK + kbase + m] = c[r] * g;
    }
    __syncthreads();

    // ---------------- phase 2: row softmax in LDS + stream attention out ----------------
#pragma unroll
    for (int rr = 0; rr < 2; ++rr) {
        const int row = wv * 2 + rr;
        float* srow = sc + row * NK;

        float mx = -3.402823466e+38f;
        for (int c0 = lane; c0 < NK; c0 += 32) mx = fmaxf(mx, srow[c0]);
#pragma unroll
        for (int off = 16; off > 0; off >>= 1) mx = fmaxf(mx, __shfl_xor(mx, off));

        float sum = 0.f;
        for (int c0 = lane; c0 < NK; c0 += 32) {
            float e = __expf(srow[c0] - mx);
            srow[c0] = e;
            sum += e;
        }
#pragma unroll
        for (int off = 16; off > 0; off >>= 1) sum += __shfl_xor(sum, off);
        const float inv = 1.0f / sum;

        float* garow = attn + ((size_t)b * NQ + q0 + row) * NK;
        for (int c0 = lane; c0 < NK; c0 += 32) {
            float p = srow[c0] * inv;
            srow[c0] = p;          // keep normalized probs for phase 3
            garow[c0] = p;         // coalesced attention write
        }
    }
    for (int i = t; i < QTILE * DH; i += 256) oacc[i] = 0.f;
    __syncthreads();

    // ---------------- phase 3: out = attn @ V via WMMA ----------------
    v8f acc0 = {}, acc1 = {}, acc2 = {}, acc3 = {};
    for (int kt = wv; kt < NK / KT3; kt += NWAVES) {
        const int kbase = kt * KT3;

        // stage V slab (32 keys x 64 dh) transposed -> [dh][key] f16
        {
            const size_t goff = ((size_t)b * NK + kbase + lane) * DH;
            const float4* v4p = (const float4*)(value + goff);
            __builtin_prefetch(value + goff + (size_t)NWAVES * KT3 * DH, 0, 1);
#pragma unroll
            for (int jj = 0; jj < 16; ++jj) {
                float4 x = v4p[jj];
                const int d = jj * 4;
                wstage[(d + 0) * KT3 + lane] = (_Float16)x.x;
                wstage[(d + 1) * KT3 + lane] = (_Float16)x.y;
                wstage[(d + 2) * KT3 + lane] = (_Float16)x.z;
                wstage[(d + 3) * KT3 + lane] = (_Float16)x.w;
            }
        }
        lds_wait();

        // A operand: probabilities row m, K slab [kbase, kbase+32)
        const float* pr = sc + m * NK + kbase;
        const v16h a = cvt2x8(pr + kb0, pr + 8 + kb0);

        // B operand: V^T, lane covers dh column nt*16+m, K contiguous
        const int koff = hi * 16;
        const v16h vb0 = *(const v16h*)(wstage + (0 * 16 + m) * KT3 + koff);
        const v16h vb1 = *(const v16h*)(wstage + (1 * 16 + m) * KT3 + koff);
        const v16h vb2 = *(const v16h*)(wstage + (2 * 16 + m) * KT3 + koff);
        const v16h vb3 = *(const v16h*)(wstage + (3 * 16 + m) * KT3 + koff);

        acc0 = __builtin_amdgcn_wmma_f32_16x16x32_f16(false, a, false, vb0, (short)0, acc0, false, false);
        acc1 = __builtin_amdgcn_wmma_f32_16x16x32_f16(false, a, false, vb1, (short)0, acc1, false, false);
        acc2 = __builtin_amdgcn_wmma_f32_16x16x32_f16(false, a, false, vb2, (short)0, acc2, false, false);
        acc3 = __builtin_amdgcn_wmma_f32_16x16x32_f16(false, a, false, vb3, (short)0, acc3, false, false);
    }

    // cross-wave reduction of partial outputs via LDS float atomics
#pragma unroll
    for (int r = 0; r < 8; ++r) {
        const int mo = (r + hi * 8) * DH;
        atomicAdd(&oacc[mo + 0 * 16 + m], acc0[r]);
        atomicAdd(&oacc[mo + 1 * 16 + m], acc1[r]);
        atomicAdd(&oacc[mo + 2 * 16 + m], acc2[r]);
        atomicAdd(&oacc[mo + 3 * 16 + m], acc3[r]);
    }
    __syncthreads();

    // coalesced output store (16x64 f32 contiguous)
    {
        float4* og = (float4*)(out + ((size_t)b * NQ + q0) * DH);
        const float4* oa = (const float4*)oacc;
        og[t] = oa[t];
    }
}

extern "C" void kernel_launch(void* const* d_in, const int* in_sizes, int n_in,
                              void* d_out, int out_size, void* d_ws, size_t ws_size,
                              hipStream_t stream) {
    (void)in_sizes; (void)n_in; (void)d_ws; (void)ws_size; (void)out_size;

    const float* query_spikes  = (const float*)d_in[0];
    const float* key_spikes    = (const float*)d_in[1];
    const float* value         = (const float*)d_in[2];
    const float* query_trace   = (const float*)d_in[3];
    const float* key_trace     = (const float*)d_in[4];
    const float* key_phases    = (const float*)d_in[5];
    const float* current_phase = (const float*)d_in[6];

    float* out  = (float*)d_out;                          // [B,NQ,DH] first
    float* attn = (float*)d_out + (size_t)BB * NQ * DH;   // then [B,NQ,NK]

    (void)hipFuncSetAttribute((const void*)spiking_attn_wmma_kernel,
                              hipFuncAttributeMaxDynamicSharedMemorySize, SMEM_TOTAL);

    dim3 grid(BB * (NQ / QTILE));   // 2048 workgroups
    dim3 block(256);                // 8 wave32 waves
    spiking_attn_wmma_kernel<<<grid, block, SMEM_TOTAL, stream>>>(
        query_spikes, key_spikes, value, query_trace, key_trace,
        key_phases, current_phase, out, attn);
}
